// MoELayer_67577015435316
// MI455X (gfx1250) — matrix operand order repaired and checked by reading, not compiled, not run on hardware
//
#include <hip/hip_runtime.h>
#include <hip/hip_bf16.h>
#include <math.h>

typedef __attribute__((ext_vector_type(16))) __bf16 v16bf;
typedef __attribute__((ext_vector_type(8)))  float  v8f;
typedef unsigned int v4u __attribute__((ext_vector_type(4)));
typedef int          v4i __attribute__((ext_vector_type(4)));
typedef int          v8i __attribute__((ext_vector_type(8)));

constexpr int Bn = 32, Cn = 256, Hn = 64, Wn = 64, En = 4;
constexpr int XROW = 65;            // padded LDS row stride (dwords) via TDM pad
constexpr int XSLAB = 32 * XROW;    // 2080 dwords per (buf,row) slab
constexpr int BROW = 20;            // act B-tile row stride in dwords (40 bf16)

static __device__ __forceinline__ unsigned short f2bf_rne(float f) {
    unsigned int u = __float_as_uint(f);
    unsigned int r = u + 0x7FFFu + ((u >> 16) & 1u);
    return (unsigned short)(r >> 16);
}

// ---- TDM: load one 2D tile (64 floats x 32 channels, chan stride 4096) into LDS,
// with 1-dword padding after every 64-dword row (LDS row stride = 65 dwords).
static __device__ __forceinline__ void tdm_load_row(const float* gptr, unsigned lds_off) {
    unsigned long long ga = (unsigned long long)(size_t)gptr;
    v4u g0;
    g0.x = 1u;                                            // count=1 user descriptor
    g0.y = lds_off;                                       // lds_addr (bytes)
    g0.z = (unsigned)(ga & 0xFFFFFFFFu);                  // global_addr[31:0]
    g0.w = (unsigned)((ga >> 32) & 0x01FFFFFFu) | (2u << 30); // addr[56:32], type=2
    v8i g1;
    g1[0] = (int)((2u << 16)      // data_size = 4 bytes
                | (1u << 20)      // pad_enable
                | (5u << 22)      // pad_interval: 64 dwords
                | (0u << 25));    // pad_amount: 1 dword
    g1[1] = (int)(64u << 16);     // tensor_dim0 = 64 (row length, elems)
    g1[2] = (int)(32u << 16);     // tensor_dim1 = 32 (channels)
    g1[3] = (int)(64u << 16);     // tile_dim0 = 64
    g1[4] = (int)(32u);           // tile_dim1 = 32, tile_dim2 = 0 (2D)
    g1[5] = (int)(4096u);         // tensor_dim0_stride = H*W elems
    g1[6] = 0;
    g1[7] = 0;
    v4i z4 = {0, 0, 0, 0};
#if __clang_major__ >= 23
    v8i z8 = {0, 0, 0, 0, 0, 0, 0, 0};
    __builtin_amdgcn_tensor_load_to_lds(g0, g1, z4, z4, z8, 0);
#else
    __builtin_amdgcn_tensor_load_to_lds(g0, g1, z4, z4, 0);
#endif
}

// ---------------- router: global average pool ----------------
__global__ void pool_kernel(const float* __restrict__ x, float* __restrict__ pooled) {
    __shared__ float red[256];
    const int bc = blockIdx.x;
    const float* p = x + (size_t)bc * (Hn * Wn);
    float s = 0.f;
    for (int i = threadIdx.x; i < Hn * Wn; i += 256) s += p[i];
    red[threadIdx.x] = s;
    __syncthreads();
    for (int off = 128; off > 0; off >>= 1) {
        if (threadIdx.x < off) red[threadIdx.x] += red[threadIdx.x + off];
        __syncthreads();
    }
    if (threadIdx.x == 0) pooled[bc] = red[0] * (1.0f / (Hn * Wn));
}

// ---------------- router: logits -> softmax/T -> top2 -> renorm ----------------
__global__ void gate_kernel(const float* __restrict__ pooled,
                            const float* __restrict__ rw,
                            const float* __restrict__ rb,
                            float* __restrict__ gate) {
    const int b = threadIdx.x;
    if (b >= Bn) return;
    const float* pr = pooled + b * Cn;
    float logit[En];
    for (int e = 0; e < En; ++e) {
        float s = rb[e];
        const float* w = rw + e * Cn;
        for (int c = 0; c < Cn; ++c) s += pr[c] * w[c];
        logit[e] = s * 0.5f;                      // /TEMP (TEMP=2)
    }
    float mx = logit[0];
    for (int e = 1; e < En; ++e) mx = fmaxf(mx, logit[e]);
    float p[En], sum = 0.f;
    for (int e = 0; e < En; ++e) { p[e] = expf(logit[e] - mx); sum += p[e]; }
    for (int e = 0; e < En; ++e) p[e] /= sum;
    int i0 = 0;
    for (int e = 1; e < En; ++e) if (p[e] > p[i0]) i0 = e;
    int i1 = -1;
    for (int e = 0; e < En; ++e) {
        if (e == i0) continue;
        if (i1 < 0 || p[e] > p[i1]) i1 = e;
    }
    const float z = p[i0] + p[i1];
    for (int e = 0; e < En; ++e)
        gate[b * En + e] = (e == i0) ? p[i0] / z : (e == i1 ? p[i1] / z : 0.0f);
}

// ---------------- pointwise weights: f32 -> bf16 ----------------
__global__ void cvt_kernel(const float* __restrict__ pw, unsigned short* __restrict__ pwbf, int n) {
    int i = blockIdx.x * blockDim.x + threadIdx.x;
    if (i < n) pwbf[i] = f2bf_rne(pw[i]);
}

// ---- fused: TDM-staged depthwise3x3 + GELU + gated bf16-WMMA 1x1 conv ----
__global__ __launch_bounds__(256) void moe_main(
        const float* __restrict__ x,
        const float* __restrict__ dw_w,
        const float* __restrict__ dw_b,
        const unsigned short* __restrict__ pwbf,
        const float* __restrict__ pw_b,
        const float* __restrict__ gate,
        float* __restrict__ out) {
    __shared__ float xs[2][3][XSLAB];              // double-buffered TDM dest, ~48.8 KB
    __shared__ unsigned short actb[64 * 2 * BROW]; // bf16 B-tile [w][c], padded rows, 5 KB

    const int h    = blockIdx.x;                 // 0..63
    const int b    = blockIdx.y;                 // 0..31
    const int tid  = threadIdx.x;
    const int lane = tid & 31;
    const int wav  = tid >> 5;                   // 8 waves
    const int half = lane >> 4;
    const int l16  = lane & 15;

    float gv[En];
    for (int e = 0; e < En; ++e) gv[e] = gate[b * En + e];

    // zero slabs for rows that TDM will never write (SAME-padding boundary)
    if (h == 0)
        for (int i = tid; i < XSLAB; i += 256) { xs[0][0][i] = 0.f; xs[1][0][i] = 0.f; }
    if (h == Hn - 1)
        for (int i = tid; i < XSLAB; i += 256) { xs[0][2][i] = 0.f; xs[1][2][i] = 0.f; }

    const unsigned xs_base = (unsigned)(size_t)&xs[0][0][0];
    const int R = ((h > 0) ? 1 : 0) + 1 + ((h < Hn - 1) ? 1 : 0); // rows per chunk

    auto stage = [&](int kc, int buf) {
        if (wav != 0) return;
        const float* gp = x + ((size_t)b * Cn + kc * 32) * (Hn * Wn);
        for (int r = 0; r < 3; ++r) {
            int hh = h - 1 + r;
            if (hh < 0 || hh >= Hn) continue;
            tdm_load_row(gp + (size_t)hh * Wn,
                         xs_base + (unsigned)(((buf * 3 + r) * XSLAB) * 4));
        }
    };

    v8f acc[8];
    for (int t = 0; t < 8; ++t)
        for (int j = 0; j < 8; ++j) acc[t][j] = 0.0f;

    stage(0, 0);                                  // prologue

    const int wb = wav * 8;                       // contiguous 8-wide w segment per wave

    for (int kc = 0; kc < Cn / 32; ++kc) {        // 8 K-chunks of 32 channels
        const int buf = kc & 1;
        if (kc < 7) stage(kc + 1, buf ^ 1);       // prefetch next chunk via TDM
        if (wav == 0) {
            if (kc < 7) {
                if (R == 3) __builtin_amdgcn_s_wait_tensorcnt(3);
                else        __builtin_amdgcn_s_wait_tensorcnt(2);
            } else {
                __builtin_amdgcn_s_wait_tensorcnt(0);
            }
        }
        __syncthreads();                          // chunk kc resident for all waves

        for (int e = 0; e < En; ++e) {
            const float g = gv[e];
            if (g == 0.0f) continue;              // uniform: skip unselected experts

            // ---- depthwise 3x3 + bias + exact GELU, scaled by gate -> bf16 LDS ----
            {
                const int c  = lane;              // one channel per lane
                const int cg = kc * 32 + c;
                const float* kw = dw_w + ((size_t)e * Cn + cg) * 9;
                const float k00 = kw[0], k01 = kw[1], k02 = kw[2];
                const float k10 = kw[3], k11 = kw[4], k12 = kw[5];
                const float k20 = kw[6], k21 = kw[7], k22 = kw[8];
                const float bb  = dw_b[e * Cn + cg];
                const float* r0 = &xs[buf][0][c * XROW + wb];
                const float* r1 = &xs[buf][1][c * XROW + wb];
                const float* r2 = &xs[buf][2][c * XROW + wb];
                // load each 10-wide row window once (never touches the pad dword)
                float v0[10], v1[10], v2[10];
                v0[0] = (wb > 0) ? r0[-1] : 0.f;
                v1[0] = (wb > 0) ? r1[-1] : 0.f;
                v2[0] = (wb > 0) ? r2[-1] : 0.f;
                #pragma unroll
                for (int i = 0; i < 8; ++i) { v0[i+1] = r0[i]; v1[i+1] = r1[i]; v2[i+1] = r2[i]; }
                v0[9] = (wb < 56) ? r0[8] : 0.f;
                v1[9] = (wb < 56) ? r1[8] : 0.f;
                v2[9] = (wb < 56) ? r2[8] : 0.f;
                #pragma unroll
                for (int q = 0; q < 8; ++q) {
                    float s = bb + v0[q] * k00 + v0[q+1] * k01 + v0[q+2] * k02
                                 + v1[q] * k10 + v1[q+1] * k11 + v1[q+2] * k12
                                 + v2[q] * k20 + v2[q+1] * k21 + v2[q+2] * k22;
                    float ge = 0.5f * s * (1.0f + erff(s * 0.70710678118654752f));
                    actb[(wb + q) * (2 * BROW) + c] = f2bf_rne(ge * g);
                }
            }
            __syncthreads();

            // ---- WMMA: A = pw_bf16[e], B = actb ----
            union Frag { unsigned int u[8]; v16bf v; };
            Frag Af[2], Bf[4];
            const unsigned int* bp = (const unsigned int*)actb;   // BROW dwords per w-row
            for (int mt = 0; mt < 2; ++mt) {
                const int m = (wav * 2 + mt) * 16 + l16;
                const unsigned int* ap =
                    (const unsigned int*)(pwbf + ((size_t)e * Cn + m) * Cn + kc * 32);
                #pragma unroll
                for (int j = 0; j < 8; ++j) {
                    // 16-bit A 16x32: VGPR j <-> K = (j<4?0:16) + 2*(j%4) + 8*half
                    int kb = ((j < 4) ? 0 : 16) + 2 * (j & 3) + 8 * half;
                    Af[mt].u[j] = ap[kb >> 1];
                }
            }
            for (int nt = 0; nt < 4; ++nt) {
                const int w = nt * 16 + l16;
                #pragma unroll
                for (int j = 0; j < 8; ++j) {
                    // 16-bit B 32x16: VGPR j <-> K = (j<4?0:16) + 2*(j%4) + 8*half, lane%16 = N
                    Bf[nt].u[j] = bp[w * BROW + ((j < 4) ? 0 : 8) + (j & 3) + 4 * half];
                }
            }
            for (int mt = 0; mt < 2; ++mt)
                for (int nt = 0; nt < 4; ++nt)
                    acc[mt * 4 + nt] = __builtin_amdgcn_wmma_f32_16x16x32_bf16(
                        false, Af[mt].v, false, Bf[nt].v,
                        (short)0, acc[mt * 4 + nt], false, false);
            __syncthreads();                      // actb reusable by next expert
        }
    }

    // ---- epilogue: gated pointwise bias + store ----
    for (int mt = 0; mt < 2; ++mt) {
        for (int j = 0; j < 8; ++j) {
            const int o = (wav * 2 + mt) * 16 + j + 8 * half; // D: VGPR j -> M = j + 8*half
            float bias = 0.f;
            for (int e = 0; e < En; ++e) bias += gv[e] * pw_b[e * Cn + o];
            for (int nt = 0; nt < 4; ++nt) {
                out[(((size_t)b * Cn + o) * Hn + h) * Wn + nt * 16 + l16] =
                    acc[mt * 4 + nt][j] + bias;
            }
        }
    }
}

extern "C" void kernel_launch(void* const* d_in, const int* in_sizes, int n_in,
                              void* d_out, int out_size, void* d_ws, size_t ws_size,
                              hipStream_t stream) {
    const float* x        = (const float*)d_in[0];
    const float* router_w = (const float*)d_in[1];
    const float* router_b = (const float*)d_in[2];
    const float* dw_w     = (const float*)d_in[3];
    const float* dw_b     = (const float*)d_in[4];
    const float* pw_w     = (const float*)d_in[5];
    const float* pw_b     = (const float*)d_in[6];
    float* out = (float*)d_out;

    char* ws = (char*)d_ws;
    float*          pooled = (float*)(ws + 0);              // 32 KB
    float*          gate   = (float*)(ws + 32768);          // 512 B
    unsigned short* pwbf   = (unsigned short*)(ws + 33280); // 512 KB

    pool_kernel<<<Bn * Cn, 256, 0, stream>>>(x, pooled);
    gate_kernel<<<1, 32, 0, stream>>>(pooled, router_w, router_b, gate);
    const int n = En * Cn * Cn;
    cvt_kernel<<<(n + 255) / 256, 256, 0, stream>>>(pw_w, pwbf, n);
    dim3 grid(Hn, Bn);
    moe_main<<<grid, 256, 0, stream>>>(x, dw_w, dw_b, pwbf, pw_b, gate, out);
}
